// ConvTransformerEncoder_48584670052875
// MI455X (gfx1250) — compile-verified, matrix operand order checked by reading, not run on hardware
//
#include <hip/hip_runtime.h>

// ---------------- problem constants ----------------
#define NB   2          // batch
#define LSEQ 192        // L
#define DM   256        // D_MODEL
#define NH   8          // heads (seq)
#define DP   128        // D_PAIR
#define DMSA 32
#define HT   4          // H_TRI
#define DKT  32         // DK_TRI

#define SEQN   (NB*LSEQ*DM)                 //  98304
#define PS     (NB*LSEQ*LSEQ*DP)            //  9437184 (pair-sized)
#define SCN    ((long)NB*LSEQ*HT*LSEQ*LSEQ) // 56623104

typedef __attribute__((ext_vector_type(8)))  float  v8f;
typedef __attribute__((ext_vector_type(4)))  float  f4;
typedef __attribute__((ext_vector_type(16))) __bf16 v16bf;

__device__ __forceinline__ float fast_rcp(float x) {
#if __has_builtin(__builtin_amdgcn_rcpf)
  return __builtin_amdgcn_rcpf(x);
#else
  return 1.f / x;
#endif
}
__device__ __forceinline__ float fast_tanh(float x) {
#if __has_builtin(__builtin_amdgcn_tanhf)
  return __builtin_amdgcn_tanhf(x);
#elif __has_builtin(__builtin_amdgcn_tanh_f32)
  return __builtin_amdgcn_tanh_f32(x);
#else
  return tanhf(x);
#endif
}
__device__ __forceinline__ float mishf(float x) {
  float sp = (x > 20.f) ? x : log1pf(__expf(x));
  return x * fast_tanh(sp);
}
__device__ __forceinline__ float sigf(float x) { return fast_rcp(1.f + __expf(-x)); }

// ---------------------------------------------------------------------------
// Fragment loaders. 16-bit A/B fragment K map: lane&15 = row/col in tile,
// lane>>4 = K-half; element e holds K = e + 8*half + (e&8).
// When the K stride is 1, a lane's elements are two contiguous 8-float runs
// -> four b128 loads instead of 16 scalar gathers. f32->bf16 via native
// casts so the compiler emits packed v_cvt_pk_bf16_f32.
// ---------------------------------------------------------------------------
template<int FAST>
__device__ __forceinline__ v16bf load_frag_row(const float* __restrict__ base,
                                               long rs, long ks, int row, int k0, int hf) {
  v16bf v;
  if (FAST) {
    const float* p = base + (long)row * rs + k0 + (hf << 3);
    f4 x0 = *(const f4*)(p);
    f4 x1 = *(const f4*)(p + 4);
    f4 x2 = *(const f4*)(p + 16);
    f4 x3 = *(const f4*)(p + 20);
#pragma unroll
    for (int e = 0; e < 4; ++e) {
      v[e]      = (__bf16)x0[e];
      v[e + 4]  = (__bf16)x1[e];
      v[e + 8]  = (__bf16)x2[e];
      v[e + 12] = (__bf16)x3[e];
    }
  } else {
#pragma unroll
    for (int e = 0; e < 16; ++e) {
      int kk = k0 + e + (hf << 3) + (e & 8);
      v[e] = (__bf16)base[(long)row * rs + (long)kk * ks];
    }
  }
  return v;
}

// ---------------------------------------------------------------------------
// Generic strided batched GEMM on WMMA bf16 -> f32.
// C[z][m,n] = act( alpha * sum_k A[z][m,k]*B[z][k,n] + bias[n] )
// One wave = one 32x32 output tile (2x2 WMMA tiles, A/B fragments reused).
// batch z decomposed: z2 = z%D2; t=z/D2; z1=t%D1; z0=t/D1.
// M%32==0, N%32==0, K%32==0.
// ---------------------------------------------------------------------------
template<int AF, int BF>
__global__ void gemm_wmma(const float* __restrict__ A, const float* __restrict__ Bm,
                          const float* __restrict__ bias, float* __restrict__ C,
                          int M, int N, int K,
                          long am, long ak, long bk, long bn, long cm, long cn,
                          int D1, int D2,
                          long a0, long a1, long a2,
                          long b0, long b1, long b2,
                          long c0, long c1, long c2,
                          int act, float alpha) {
  int lane = threadIdx.x;
  int tn0 = blockIdx.x * 32;
  int tm0 = blockIdx.y * 32;
  int z  = blockIdx.z;
  int z2 = z % D2; int t = z / D2; int z1 = t % D1; int z0 = t / D1;
  const float* Ab = A  + (long)z0 * a0 + (long)z1 * a1 + (long)z2 * a2;
  const float* Bb = Bm + (long)z0 * b0 + (long)z1 * b1 + (long)z2 * b2;
  float*       Cb = C  + (long)z0 * c0 + (long)z1 * c1 + (long)z2 * c2;

  int hf = lane >> 4;        // K-half selector
  int mr = lane & 15;        // row (A) / col (B,C) within 16-tile

  v8f acc00 = {}, acc01 = {}, acc10 = {}, acc11 = {};
#pragma unroll 2
  for (int k0 = 0; k0 < K; k0 += 32) {
    if (k0 + 32 < K) {   // prefetch next K block (gfx1250 global_prefetch)
      __builtin_prefetch(Ab + (long)(tm0 + mr) * am + (long)(k0 + 32) * ak, 0, 3);
      __builtin_prefetch(Bb + (long)(k0 + 32) * bk + (long)(tn0 + mr) * bn, 0, 3);
    }
    v16bf a0  = load_frag_row<AF>(Ab, am, ak, tm0 + mr,      k0, hf);
    v16bf a1  = load_frag_row<AF>(Ab, am, ak, tm0 + 16 + mr, k0, hf);
    v16bf bb0 = load_frag_row<BF>(Bb, bn, bk, tn0 + mr,      k0, hf);
    v16bf bb1 = load_frag_row<BF>(Bb, bn, bk, tn0 + 16 + mr, k0, hf);
    acc00 = __builtin_amdgcn_wmma_f32_16x16x32_bf16(false, a0, false, bb0, (short)0, acc00, false, false);
    acc01 = __builtin_amdgcn_wmma_f32_16x16x32_bf16(false, a0, false, bb1, (short)0, acc01, false, false);
    acc10 = __builtin_amdgcn_wmma_f32_16x16x32_bf16(false, a1, false, bb0, (short)0, acc10, false, false);
    acc11 = __builtin_amdgcn_wmma_f32_16x16x32_bf16(false, a1, false, bb1, (short)0, acc11, false, false);
  }
#pragma unroll
  for (int sm_ = 0; sm_ < 2; ++sm_) {
#pragma unroll
    for (int sn_ = 0; sn_ < 2; ++sn_) {
      v8f acc = sm_ == 0 ? (sn_ == 0 ? acc00 : acc01) : (sn_ == 0 ? acc10 : acc11);
      int n = tn0 + sn_ * 16 + mr;
#pragma unroll
      for (int r = 0; r < 8; ++r) {
        int m = tm0 + sm_ * 16 + r + (hf << 3);
        float v = acc[r] * alpha;
        if (bias) v += bias[n];
        if (act == 1) v = fmaxf(v, 0.f);
        else if (act == 2) v = sigf(v);
        Cb[(long)m * cm + (long)n * cn] = v;
      }
    }
  }
}

// ---------------- elementwise / reduction kernels ----------------
__global__ void k_mask_seq(const float* s, const float* mask, float* o) {
  long i = (long)blockIdx.x * blockDim.x + threadIdx.x;
  if (i >= SEQN) return;
  long bl = i / DM;
  o[i] = s[i] * mask[bl];
}

__global__ void k_conv_mish(const float* sb, const float* w, const float* cb, float* o) {
  long idx = (long)blockIdx.x * blockDim.x + threadIdx.x;
  if (idx >= SEQN) return;
  int oc = idx % DM; int l = (idx / DM) % LSEQ; int b = idx / (DM * LSEQ);
  float acc = cb[oc];
  for (int dk = 0; dk < 3; ++dk) {
    int ll = l + dk - 1;
    if (ll < 0 || ll >= LSEQ) continue;
    const float* xr = sb + ((long)(b * LSEQ + ll)) * DM;
    const float* wr = w + (long)oc * DM * 3 + dk;
    for (int i = 0; i < DM; ++i) acc += xr[i] * wr[i * 3];
  }
  o[idx] = sb[idx] + mishf(acc);
}

__global__ void k_ln(const float* x, float* y, const float* g, const float* b, int W) {
  __shared__ float red[256];
  int row = blockIdx.x, t = threadIdx.x;
  float v = x[(long)row * W + t];
  red[t] = v; __syncthreads();
  for (int s = blockDim.x >> 1; s > 0; s >>= 1) { if (t < s) red[t] += red[t + s]; __syncthreads(); }
  float mean = red[0] / W; __syncthreads();
  float d = v - mean;
  red[t] = d * d; __syncthreads();
  for (int s = blockDim.x >> 1; s > 0; s >>= 1) { if (t < s) red[t] += red[t + s]; __syncthreads(); }
  float var = red[0] / W;
  y[(long)row * W + t] = d * rsqrtf(var + 1e-5f) * g[t] + b[t];
}

__global__ void k_p2h(const float* pn, const float* w, float* bt) {   // (b,h,i,j)
  long idx = (long)blockIdx.x * blockDim.x + threadIdx.x;
  long tot = (long)NB * NH * LSEQ * LSEQ;
  if (idx >= tot) return;
  int j = idx % LSEQ; long t = idx / LSEQ;
  int i = t % LSEQ; t /= LSEQ;
  int h = t % NH;   int b = t / NH;
  const float* row = pn + ((long)(b * LSEQ + i) * LSEQ + j) * DP;
  float acc = 0.f;
  for (int p = 0; p < DP; ++p) acc += row[p] * w[p * NH + h];
  bt[idx] = acc;
}

__global__ void k_bbk(const float* zn, const float* w, float* bb) {   // ((b,i,j),h)
  long idx = (long)blockIdx.x * blockDim.x + threadIdx.x;
  long tot = (long)NB * LSEQ * LSEQ * HT;
  if (idx >= tot) return;
  long row = idx / HT; int h = idx % HT;
  const float* zr = zn + row * DP;
  float acc = 0.f;
  for (int p = 0; p < DP; ++p) acc += zr[p] * w[p * HT + h];
  bb[idx] = acc;
}

__global__ void k_softmax_seq(const float* sc, const float* bt, const float* mask, float* attn) {
  __shared__ float red[256];
  int row = blockIdx.x;                 // (b,h,i)
  int j = threadIdx.x;
  int b = row / (NH * LSEQ);
  float v = -1e30f;
  if (j < LSEQ) {
    float s = sc[(long)row * LSEQ + j] + bt[(long)row * LSEQ + j];
    v = (mask[b * LSEQ + j] > 0.f) ? s : -1e9f;
  }
  red[j] = v; __syncthreads();
  for (int s = 128; s > 0; s >>= 1) { if (j < s) red[j] = fmaxf(red[j], red[j + s]); __syncthreads(); }
  float mx = red[0]; __syncthreads();
  float e = (j < LSEQ) ? __expf(v - mx) : 0.f;
  red[j] = e; __syncthreads();
  for (int s = 128; s > 0; s >>= 1) { if (j < s) red[j] += red[j + s]; __syncthreads(); }
  if (j < LSEQ) attn[(long)row * LSEQ + j] = e * fast_rcp(red[0]);
}

__global__ void k_softmax_tri(float* sc, const float* bb, const float* mask) {
  __shared__ float red[256];
  long row = blockIdx.x;                // ((b,l1,h,i))
  int j = threadIdx.x;
  int i = row % LSEQ; long t = row / LSEQ;
  int h = t % HT; t /= HT;
  int b = (int)(t / LSEQ);
  long base = row * LSEQ;
  float v = -1e30f;
  if (j < LSEQ) {
    float s = sc[base + j] + bb[(((long)(b * LSEQ + i)) * LSEQ + j) * HT + h];
    v = (mask[b * LSEQ + j] > 0.f) ? s : -1e9f;
  }
  red[j] = v; __syncthreads();
  for (int s = 128; s > 0; s >>= 1) { if (j < s) red[j] = fmaxf(red[j], red[j + s]); __syncthreads(); }
  float mx = red[0]; __syncthreads();
  float e = (j < LSEQ) ? __expf(v - mx) : 0.f;
  red[j] = e; __syncthreads();
  for (int s = 128; s > 0; s >>= 1) { if (j < s) red[j] += red[j + s]; __syncthreads(); }
  if (j < LSEQ) sc[base + j] = e * fast_rcp(red[0]);
}

__global__ void k_add_mish(const float* a, const float* r, float* o, long n) {
  long i = (long)blockIdx.x * blockDim.x + threadIdx.x;
  if (i < n) o[i] = a[i] + mishf(r[i]);
}
__global__ void k_add(const float* a, const float* r, float* o, long n) {
  long i = (long)blockIdx.x * blockDim.x + threadIdx.x;
  if (i < n) o[i] = a[i] + r[i];
}
__global__ void k_add_ip(float* z, const float* r, long n) {
  long i = (long)blockIdx.x * blockDim.x + threadIdx.x;
  if (i < n) z[i] += r[i];
}
__global__ void k_mul_ip(float* a, const float* g, long n) {
  long i = (long)blockIdx.x * blockDim.x + threadIdx.x;
  if (i < n) a[i] *= g[i];
}
__global__ void k_gated_add(float* z, const float* p, const float* g, long n) {
  long i = (long)blockIdx.x * blockDim.x + threadIdx.x;
  if (i < n) z[i] += p[i] * g[i];
}
__global__ void k_pair_add(const float* pair, const float* o, const float* bias, float* z) {
  long i = (long)blockIdx.x * blockDim.x + threadIdx.x;
  if (i >= PS) return;
  z[i] = pair[i] + o[i] + bias[i % DP];
}
__global__ void k_trimask(float* x, const float* gate, const float* mask) {
  long i = (long)blockIdx.x * blockDim.x + threadIdx.x;
  if (i >= PS) return;
  long t = i / DP;
  int j = t % LSEQ; t /= LSEQ;
  int ii = t % LSEQ; int b = t / LSEQ;
  x[i] = x[i] * gate[i] * mask[b * LSEQ + ii] * mask[b * LSEQ + j];
}
__global__ void k_tw2(const float* w2, float* wt) {   // (d,e,p) -> (e,(d,p))
  long i = (long)blockIdx.x * blockDim.x + threadIdx.x;
  if (i >= (long)DMSA * DMSA * DP) return;
  int p = i % DP; int e = (i / DP) % DMSA; int d = i / (DP * DMSA);
  wt[((long)e * DMSA + d) * DP + p] = w2[i];
}
__global__ void k_repackU(const float* uc, float* up) { // (b,j,(d,p)) -> (b,d,(j,p))
  long i = (long)blockIdx.x * blockDim.x + threadIdx.x;
  long tot = (long)NB * LSEQ * DMSA * DP;
  if (i >= tot) return;
  int p = i % DP; long t = i / DP;
  int d = t % DMSA; t /= DMSA;
  int j = t % LSEQ; int b = t / LSEQ;
  up[(((long)(b * DMSA + d)) * LSEQ + j) * DP + p] = uc[i];
}

// ---------------- host-side helpers ----------------
static inline void gemm(hipStream_t st, const float* A, const float* B, const float* bias, float* C,
                        int M, int N, int K,
                        long am, long ak, long bk, long bn, long cm, long cn,
                        int nz, int D1, int D2,
                        long a0, long a1, long a2, long b0, long b1, long b2,
                        long c0, long c1, long c2, int act, float alpha) {
  dim3 g(N / 32, M / 32, nz);
  if (ak == 1 && bk == 1)
    gemm_wmma<1,1><<<g, 32, 0, st>>>(A, B, bias, C, M, N, K, am, ak, bk, bn, cm, cn,
                                     D1, D2, a0, a1, a2, b0, b1, b2, c0, c1, c2, act, alpha);
  else if (ak == 1)
    gemm_wmma<1,0><<<g, 32, 0, st>>>(A, B, bias, C, M, N, K, am, ak, bk, bn, cm, cn,
                                     D1, D2, a0, a1, a2, b0, b1, b2, c0, c1, c2, act, alpha);
  else if (bk == 1)
    gemm_wmma<0,1><<<g, 32, 0, st>>>(A, B, bias, C, M, N, K, am, ak, bk, bn, cm, cn,
                                     D1, D2, a0, a1, a2, b0, b1, b2, c0, c1, c2, act, alpha);
  else
    gemm_wmma<0,0><<<g, 32, 0, st>>>(A, B, bias, C, M, N, K, am, ak, bk, bn, cm, cn,
                                     D1, D2, a0, a1, a2, b0, b1, b2, c0, c1, c2, act, alpha);
}
static inline dim3 g1(long n) { return dim3((unsigned)((n + 255) / 256)); }

extern "C" void kernel_launch(void* const* d_in, const int* in_sizes, int n_in,
                              void* d_out, int out_size, void* d_ws, size_t ws_size,
                              hipStream_t stream) {
  (void)in_sizes; (void)n_in; (void)out_size; (void)ws_size;
  const float* seq  = (const float*)d_in[0];
  const float* pair = (const float*)d_in[1];
  const float* mask = (const float*)d_in[2];
  int pi = 3;
  auto P = [&](void) { return (const float*)d_in[pi++]; };
  const float *conv_w=P(), *conv_b=P();
  const float *n1g=P(), *n1b=P(), *n2g=P(), *n2b=P(), *n3g=P(), *n3b=P();
  const float *png=P(), *pnb=P(), *p2h=P();
  const float *wq=P(), *wqb=P(), *wk=P(), *wkb=P(), *wv=P(), *wvb=P();
  const float *fcw=P(), *fcb=P(), *s1w=P(), *s1b=P(), *s2w=P(), *s2b=P();
  const float *o1w=P(), *o1b=P(), *o2w=P(), *o2b=P();
  struct TM { const float *ng,*nb,*lw,*lb,*rw,*rb,*lgw,*lgb,*rgw,*rgb,*ogw,*ogb,*ong,*onb,*ow,*ob; } tm[2];
  for (int t = 0; t < 2; ++t)
    tm[t] = TM{P(),P(),P(),P(),P(),P(),P(),P(),P(),P(),P(),P(),P(),P(),P(),P()};
  struct TA { const float *ng,*nb,*qkv,*pw,*gw,*gb,*ow,*ob; } ta[2];
  for (int t = 0; t < 2; ++t) ta[t] = TA{P(),P(),P(),P(),P(),P(),P(),P()};
  const float *ptg=P(), *ptb=P(), *pt1w=P(), *pt1b=P(), *pt2w=P(), *pt2b=P();

  // ---- workspace layout (floats) ----
  float* ws = (float*)d_ws;
  float* Z   = ws;
  float* S1  = ws + (long)1 * PS;
  float* S2  = ws + (long)2 * PS;
  float* S3  = ws + (long)3 * PS;
  float* S4  = ws + (long)4 * PS;
  float* S5  = ws + (long)5 * PS;
  float* S6  = ws + (long)6 * PS;
  float* SC  = ws + (long)7 * PS;           // 56.6M floats
  float* sm  = SC + SCN;
  float* sbuf = sm;          float* tmp  = sm + SEQN;     float* src0 = sm + 2*SEQN;
  float* qb   = sm + 3*SEQN; float* kb   = sm + 4*SEQN;   float* vb   = sm + 5*SEQN;
  float* ctx  = sm + 6*SEQN; float* resb = sm + 7*SEQN;
  float* ffh  = sm + 8*SEQN;                               // 384*1024
  float* scS  = ffh + 393216;                              // B*H*L*L
  float* biasT= scS + 589824;
  float* BBb  = biasT + 589824;                            // B*L*L*H
  float* sS   = BBb + 294912;                              // 384*32
  float* Wt   = sS + 12288;                                // 32*4096
  float* Uc   = Wt + 131072;                               // 384*4096
  float* Up   = Uc + 1572864;                              // B*32*L*128

  float* out_src  = (float*)d_out;
  float* out_z    = out_src + SEQN;
  float* out_attn = out_z + PS;

  const float inv32 = 0.17677669529663687f;   // 1/sqrt(32)
  const long LL128 = (long)LSEQ * DP, LLL128 = (long)LSEQ * LSEQ * DP;
  const long SCb = (long)LSEQ * HT * LSEQ * LSEQ, SCl = (long)HT * LSEQ * LSEQ, SCh = (long)LSEQ * LSEQ;

  // ================= sequence path =================
  k_mask_seq<<<g1(SEQN), 256, 0, stream>>>(seq, mask, sbuf);
  k_conv_mish<<<g1(SEQN), 256, 0, stream>>>(sbuf, conv_w, conv_b, tmp);
  k_ln<<<NB * LSEQ, 256, 0, stream>>>(tmp, src0, n3g, n3b, DM);
  // pair LN for attention bias
  k_ln<<<NB * LSEQ * LSEQ, 128, 0, stream>>>(pair, S1, png, pnb, DP);
  k_p2h<<<g1((long)NB * NH * LSEQ * LSEQ), 256, 0, stream>>>(S1, p2h, biasT);
  // q,k,v
  gemm(stream, src0, wq, wqb, qb, 384, 256, 256, 256,1, 256,1, 256,1, 1,1,1, 0,0,0, 0,0,0, 0,0,0, 0, 1.f);
  gemm(stream, src0, wk, wkb, kb, 384, 256, 256, 256,1, 256,1, 256,1, 1,1,1, 0,0,0, 0,0,0, 0,0,0, 0, 1.f);
  gemm(stream, src0, wv, wvb, vb, 384, 256, 256, 256,1, 256,1, 256,1, 1,1,1, 0,0,0, 0,0,0, 0,0,0, 0, 1.f);
  // scores (b,h,i,j) = q . k^T / sqrt(dk)
  gemm(stream, qb, kb, nullptr, scS, LSEQ, LSEQ, 32, 256,1, 1,256, LSEQ,1,
       NB*NH, 1, NH, (long)LSEQ*256,0,32, (long)LSEQ*256,0,32, (long)NH*LSEQ*LSEQ,0,(long)LSEQ*LSEQ, 0, inv32);
  k_softmax_seq<<<NB * NH * LSEQ, 256, 0, stream>>>(scS, biasT, mask, out_attn);
  // ctx (b,i,h,d) = attn @ v
  gemm(stream, out_attn, vb, nullptr, ctx, LSEQ, 32, LSEQ, LSEQ,1, 256,1, 256,1,
       NB*NH, 1, NH, (long)NH*LSEQ*LSEQ,0,(long)LSEQ*LSEQ, (long)LSEQ*256,0,32, (long)LSEQ*256,0,32, 0, 1.f);
  gemm(stream, ctx, fcw, fcb, resb, 384, 256, 256, 256,1, 256,1, 256,1, 1,1,1, 0,0,0, 0,0,0, 0,0,0, 0, 1.f);
  k_add_mish<<<g1(SEQN), 256, 0, stream>>>(src0, resb, tmp, SEQN);
  k_ln<<<NB * LSEQ, 256, 0, stream>>>(tmp, src0, n1g, n1b, DM);
  gemm(stream, src0, s1w, s1b, ffh, 384, 1024, 256, 256,1, 1024,1, 1024,1, 1,1,1, 0,0,0, 0,0,0, 0,0,0, 1, 1.f);
  gemm(stream, ffh, s2w, s2b, resb, 384, 256, 1024, 1024,1, 256,1, 256,1, 1,1,1, 0,0,0, 0,0,0, 0,0,0, 1, 1.f);
  k_add<<<g1(SEQN), 256, 0, stream>>>(src0, resb, tmp, SEQN);
  k_ln<<<NB * LSEQ, 256, 0, stream>>>(tmp, out_src, n2g, n2b, DM);   // final src -> d_out

  // ================= outer product mean =================
  gemm(stream, out_src, o1w, o1b, sS, 384, 32, 256, 256,1, 32,1, 32,1, 1,1,1, 0,0,0, 0,0,0, 0,0,0, 0, 1.f);
  k_tw2<<<g1((long)DMSA * DMSA * DP), 256, 0, stream>>>(o2w, Wt);
  gemm(stream, sS, Wt, nullptr, Uc, 384, 4096, 32, 32,1, 4096,1, 4096,1, 1,1,1, 0,0,0, 0,0,0, 0,0,0, 0, 1.f);
  k_repackU<<<g1((long)NB * LSEQ * DMSA * DP), 256, 0, stream>>>(Uc, Up);
  gemm(stream, sS, Up, nullptr, S2, LSEQ, LSEQ * DP, 32, 32,1, LL128,1, LL128,1,
       NB, 1, 1, (long)LSEQ*32,0,0, (long)DMSA*LL128,0,0, (long)LSEQ*LL128,0,0, 0, 1.f);
  k_pair_add<<<g1(PS), 256, 0, stream>>>(pair, S2, o2b, Z);

  // ================= triangle multiplications =================
  for (int t = 0; t < 2; ++t) {
    const TM& m = tm[t];
    int outgoing = (t == 0);
    k_ln<<<NB * LSEQ * LSEQ, 128, 0, stream>>>(Z, S1, m.ng, m.nb, DP);
    gemm(stream, S1, m.lw,  m.lb,  S3, PS/DP, DP, DP, DP,1, DP,1, DP,1, 1,1,1, 0,0,0, 0,0,0, 0,0,0, 0, 1.f);
    gemm(stream, S1, m.lgw, m.lgb, S4, PS/DP, DP, DP, DP,1, DP,1, DP,1, 1,1,1, 0,0,0, 0,0,0, 0,0,0, 2, 1.f);
    k_trimask<<<g1(PS), 256, 0, stream>>>(S3, S4, mask);          // left in S3
    gemm(stream, S1, m.rw,  m.rb,  S4, PS/DP, DP, DP, DP,1, DP,1, DP,1, 1,1,1, 0,0,0, 0,0,0, 0,0,0, 0, 1.f);
    gemm(stream, S1, m.rgw, m.rgb, S5, PS/DP, DP, DP, DP,1, DP,1, DP,1, 1,1,1, 0,0,0, 0,0,0, 0,0,0, 2, 1.f);
    k_trimask<<<g1(PS), 256, 0, stream>>>(S4, S5, mask);          // right in S4
    gemm(stream, S1, m.ogw, m.ogb, S5, PS/DP, DP, DP, DP,1, DP,1, DP,1, 1,1,1, 0,0,0, 0,0,0, 0,0,0, 2, 1.f);
    if (outgoing)   // out[b,i,j,d] = sum_k L[b,i,k,d] R[b,j,k,d], batched over (b,d)
      gemm(stream, S3, S4, nullptr, S6, LSEQ, LSEQ, LSEQ, LL128,DP, DP,LL128, LL128,DP,
           NB*DP, 1, DP, LLL128,0,1, LLL128,0,1, LLL128,0,1, 0, 1.f);
    else            // out[b,i,j,d] = sum_k L[b,k,i,d] R[b,k,j,d]
      gemm(stream, S3, S4, nullptr, S6, LSEQ, LSEQ, LSEQ, DP,LL128, LL128,DP, LL128,DP,
           NB*DP, 1, DP, LLL128,0,1, LLL128,0,1, LLL128,0,1, 0, 1.f);
    k_ln<<<NB * LSEQ * LSEQ, 128, 0, stream>>>(S6, S2, m.ong, m.onb, DP);
    k_mul_ip<<<g1(PS), 256, 0, stream>>>(S2, S5, PS);
    gemm(stream, S2, m.ow, m.ob, S3, PS/DP, DP, DP, DP,1, DP,1, DP,1, 1,1,1, 0,0,0, 0,0,0, 0,0,0, 0, 1.f);
    k_add_ip<<<g1(PS), 256, 0, stream>>>(Z, S3, PS);
  }

  // ================= triangle attentions =================
  for (int t = 0; t < 2; ++t) {
    const TA& a = ta[t];
    int rowm = (t == 0);
    k_ln<<<NB * LSEQ * LSEQ, 128, 0, stream>>>(Z, S1, a.ng, a.nb, DP);
    gemm(stream, S1, a.qkv + 0,   nullptr, S2, PS/DP, DP, DP, DP,1, 384,1, DP,1, 1,1,1, 0,0,0, 0,0,0, 0,0,0, 0, 1.f);
    gemm(stream, S1, a.qkv + 128, nullptr, S3, PS/DP, DP, DP, DP,1, 384,1, DP,1, 1,1,1, 0,0,0, 0,0,0, 0,0,0, 0, 1.f);
    gemm(stream, S1, a.qkv + 256, nullptr, S4, PS/DP, DP, DP, DP,1, 384,1, DP,1, 1,1,1, 0,0,0, 0,0,0, 0,0,0, 0, 1.f);
    k_bbk<<<g1((long)NB * LSEQ * LSEQ * HT), 256, 0, stream>>>(S1, a.pw, BBb);
    gemm(stream, S1, a.gw, a.gb, S5, PS/DP, DP, DP, DP,1, DP,1, DP,1, 1,1,1, 0,0,0, 0,0,0, 0,0,0, 2, 1.f);
    if (rowm) {     // batch (b,r,h): scores(i,j) over K=32
      gemm(stream, S2, S3, nullptr, SC, LSEQ, LSEQ, 32, DP,1, 1,DP, LSEQ,1,
           NB*LSEQ*HT, LSEQ, HT, LLL128,LL128,32, LLL128,LL128,32, SCb,SCl,SCh, 0, inv32);
      k_softmax_tri<<<NB * LSEQ * HT * LSEQ, 256, 0, stream>>>(SC, BBb, mask);
      gemm(stream, SC, S4, nullptr, S6, LSEQ, 32, LSEQ, LSEQ,1, DP,1, DP,1,
           NB*LSEQ*HT, LSEQ, HT, SCb,SCl,SCh, LLL128,LL128,32, LLL128,LL128,32, 0, 1.f);
    } else {        // batch (b,l,h): scores(i,j) over K=32, i/j stride L*128
      gemm(stream, S2, S3, nullptr, SC, LSEQ, LSEQ, 32, LL128,1, 1,LL128, LSEQ,1,
           NB*LSEQ*HT, LSEQ, HT, LLL128,DP,32, LLL128,DP,32, SCb,SCl,SCh, 0, inv32);
      k_softmax_tri<<<NB * LSEQ * HT * LSEQ, 256, 0, stream>>>(SC, BBb, mask);
      gemm(stream, SC, S4, nullptr, S6, LSEQ, 32, LSEQ, LSEQ,1, LL128,1, LL128,1,
           NB*LSEQ*HT, LSEQ, HT, SCb,SCl,SCh, LLL128,DP,32, LLL128,DP,32, 0, 1.f);
    }
    gemm(stream, S6, a.ow, a.ob, S1, PS/DP, DP, DP, DP,1, DP,1, DP,1, 1,1,1, 0,0,0, 0,0,0, 0,0,0, 0, 1.f);
    k_gated_add<<<g1(PS), 256, 0, stream>>>(Z, S1, S5, PS);
  }

  // ================= pair transition =================
  k_ln<<<NB * LSEQ * LSEQ, 128, 0, stream>>>(Z, S1, ptg, ptb, DP);
  gemm(stream, S1, pt1w, pt1b, SC, PS/DP, 512, DP, DP,1, 512,1, 512,1, 1,1,1, 0,0,0, 0,0,0, 0,0,0, 1, 1.f);
  gemm(stream, SC, pt2w, pt2b, S2, PS/DP, DP, 512, 512,1, DP,1, DP,1, 1,1,1, 0,0,0, 0,0,0, 0,0,0, 1, 1.f);
  k_add<<<g1(PS), 256, 0, stream>>>(Z, S2, out_z, PS);   // final z -> d_out
}
